// MCL_SHIFTGCN_47175920780080
// MI455X (gfx1250) — matrix operand order repaired and checked by reading, not compiled, not run on hardware
//
#include <hip/hip_runtime.h>
#include <hip/hip_bf16.h>

// ---------------------------------------------------------------------------
// Shapes
#define B_   64
#define C_   256
#define T_   75
#define V_   25
#define HID_ 256
#define CIN_ 512            // C + HID
#define CO_  3072           // 4*HID*K
#define KSP_ 3

typedef __attribute__((ext_vector_type(16))) __bf16          v16bf;
typedef __attribute__((ext_vector_type(8)))  float           v8f;
typedef __attribute__((ext_vector_type(16))) unsigned short  v16u;
typedef __attribute__((ext_vector_type(8)))  unsigned short  v8u;

__device__ __forceinline__ unsigned short f2bf(float f) {
    unsigned int u = __builtin_bit_cast(unsigned int, f);
    unsigned int r = u + 0x7FFFu + ((u >> 16) & 1u);   // RNE
    return (unsigned short)(r >> 16);
}

// ---------------------------------------------------------------------------
// 1) Per-(view,batch) LayerNorm statistics over C*T*V = 480000 elements.
__global__ __launch_bounds__(256) void ln_stats_kernel(
    const float* __restrict__ x1, const float* __restrict__ x2,
    const float* __restrict__ x3, float* __restrict__ stats)
{
    __shared__ float ssum[256], ssq[256];
    const int s = blockIdx.x / B_;
    const int b = blockIdx.x % B_;
    const float* x = (s == 0 ? x1 : (s == 1 ? x2 : x3)) + (size_t)b * (C_*T_*V_);
    float sum = 0.f, sq = 0.f;
    for (int i = threadIdx.x; i < C_*T_*V_; i += 256) {
        float v = x[i]; sum += v; sq += v * v;
    }
    ssum[threadIdx.x] = sum; ssq[threadIdx.x] = sq;
    __syncthreads();
    for (int off = 128; off > 0; off >>= 1) {
        if (threadIdx.x < off) {
            ssum[threadIdx.x] += ssum[threadIdx.x + off];
            ssq [threadIdx.x] += ssq [threadIdx.x + off];
        }
        __syncthreads();
    }
    if (threadIdx.x == 0) {
        float mean = ssum[0] * (1.f / (C_*T_*V_));
        float var  = ssq[0] * (1.f / (C_*T_*V_)) - mean * mean;
        stats[2*blockIdx.x]     = mean;
        stats[2*blockIdx.x + 1] = rsqrtf(var + 1e-5f);
    }
}

// ---------------------------------------------------------------------------
// 2) fm[s][v][c] = tanh(mask_s[v][c]) + 1
__global__ void mask_factor_kernel(const float* __restrict__ m1,
                                   const float* __restrict__ m2,
                                   const float* __restrict__ m3,
                                   float* __restrict__ fm)
{
    int i = blockIdx.x * 256 + threadIdx.x;
    if (i < V_*C_) {
        fm[i]            = tanhf(m1[i]) + 1.f;
        fm[V_*C_ + i]    = tanhf(m2[i]) + 1.f;
        fm[2*V_*C_ + i]  = tanhf(m3[i]) + 1.f;
    }
}

// ---------------------------------------------------------------------------
// 3) y[b][t][v][c] = max_s( (x_s - mu)*rstd * fm_s[v][c] ), stored bf16.
__global__ __launch_bounds__(256) void build_y_kernel(
    const float* __restrict__ x1, const float* __restrict__ x2,
    const float* __restrict__ x3, const float* __restrict__ stats,
    const float* __restrict__ fm, unsigned short* __restrict__ ybf)
{
    __shared__ float ytile[C_*V_];   // [c][v]
    const int b = blockIdx.x / T_;
    const int t = blockIdx.x % T_;
    for (int i = threadIdx.x; i < C_*V_; i += 256) ytile[i] = -3.4e38f;
    __syncthreads();
    for (int s = 0; s < 3; ++s) {
        const float* x = (s == 0 ? x1 : (s == 1 ? x2 : x3));
        const float mean = stats[2*(s*B_ + b)];
        const float rstd = stats[2*(s*B_ + b) + 1];
        for (int i = threadIdx.x; i < C_*V_; i += 256) {
            int c = i / V_, v = i % V_;
            float val = (x[(((size_t)b*C_ + c)*T_ + t)*V_ + v] - mean) * rstd;
            val *= fm[s*(V_*C_) + v*C_ + c];
            float cur = ytile[i];
            ytile[i] = val > cur ? val : cur;
        }
    }
    __syncthreads();
    for (int i = threadIdx.x; i < C_*V_; i += 256) {
        int v = i / C_, c = i % C_;
        ybf[((size_t)blockIdx.x * V_ + v)*C_ + c] = f2bf(ytile[c*V_ + v]);
    }
}

// ---------------------------------------------------------------------------
// 4) Weight permute+cast: wbf[j][co][cin] = bf16(conv_w[co][cin][j])
__global__ void wbf_kernel(const float* __restrict__ w,
                           unsigned short* __restrict__ wbf)
{
    size_t i = (size_t)blockIdx.x * 256 + threadIdx.x;
    if (i < (size_t)KSP_*CO_*CIN_) {
        int j   = (int)(i / (CO_*CIN_));
        int rem = (int)(i % (CO_*CIN_));
        int co  = rem / CIN_, cin = rem % CIN_;
        wbf[i] = f2bf(w[((size_t)co*CIN_ + cin)*KSP_ + j]);
    }
}

// ---------------------------------------------------------------------------
// 5) A-folded bias: biasA[c4][w] = sum_{k,v} A[k,v,w] * conv_b[k*1024 + c4]
__global__ void biasA_kernel(const float* __restrict__ A,
                             const float* __restrict__ bias,
                             float* __restrict__ biasA)
{
    int i = blockIdx.x * 256 + threadIdx.x;
    if (i < 4*HID_*V_) {
        int c4 = i / V_, w = i % V_;
        float s = 0.f;
        for (int k = 0; k < KSP_; ++k) {
            float bk = bias[k*(4*HID_) + c4];
            for (int v = 0; v < V_; ++v)
                s += A[(k*V_ + v)*V_ + w] * bk;
        }
        biasA[i] = s;
    }
}

// ---------------------------------------------------------------------------
// 6) zero h0 (bf16) and c0 (f32)
__global__ void init_state_kernel(unsigned short* __restrict__ hbf,
                                  float* __restrict__ cstate)
{
    int i = blockIdx.x * 256 + threadIdx.x;
    if (i < B_*HID_*V_) { hbf[i] = 0; cstate[i] = 0.f; }
}

// ---------------------------------------------------------------------------
// 7) Per-step conv GEMM via bf16 WMMA.
//    Zc[b][co][v] = sum_{j,cin} W[j][co][cin] * comb[b][cin][v+j-1]
//    Block: M=128 co, one batch, 32 padded v columns. 8 waves (4M x 2N).
//    W tile (128 x 64 K) is ping-pong double-buffered: one barrier / 64-K step,
//    all staging through 16B vector loads + b128 LDS stores.
__global__ __launch_bounds__(256) void gemm_step_kernel(
    const unsigned short* __restrict__ ybf,
    const unsigned short* __restrict__ hbf,
    const unsigned short* __restrict__ wbf,
    float* __restrict__ zc, int t)
{
    __shared__ unsigned short Wl[2][128*64];   // ping-pong, 16 KB each
    __shared__ unsigned short Xl[34*512];      // [vv][cin], zero halo, 34.8 KB
    const int tid  = threadIdx.x;
    const int lane = tid & 31;
    const int wave = tid >> 5;
    const int wm   = wave & 3;          // M offset = 32*wm
    const int wn   = wave >> 2;         // N offset = 16*wn
    const int co0  = blockIdx.x * 128;
    const int b    = blockIdx.y;

    // ---- Stage comb = [y ; h] with zero halo (vectorized 16B) -------------
    for (int i = tid; i < (34*512)/8; i += 256) {
        int e = i * 8;
        int vv = e >> 9, cin = e & 511, v = vv - 1;
        v8u val = {0, 0, 0, 0, 0, 0, 0, 0};
        if (v >= 0 && v < V_) {
            const unsigned short* src =
                (cin < C_) ? &ybf[((size_t)(b*T_ + t)*V_ + v)*C_ + cin]
                           : &hbf[((size_t)b*V_ + v)*HID_ + (cin - C_)];
            val = *(const v8u*)src;
        }
        *(v8u*)&Xl[e] = val;
    }

    // ---- Prime W buffer 0 (j=0, kc=0, 64 K) -------------------------------
    for (int i = tid; i < (128*64)/8; i += 256) {
        int e = i * 8;
        int row = e >> 6, cc = e & 63;
        *(v8u*)&Wl[0][row*64 + cc] =
            *(const v8u*)&wbf[((size_t)co0 + row)*CIN_ + cc];
    }
    __syncthreads();

    v8f acc0 = {}, acc1 = {};
    const int ahalf = lane >> 4;                 // 0/1
    const int arow  = wm*32 + (lane & 15);       // A row (ms adds +16)
    const int bcol  = wn*16 + (lane & 15);       // padded v column 0..31

    const int NIT = KSP_ * (CIN_/64);            // 24 pipeline steps
    int buf = 0;
    for (int it = 0; it < NIT; ++it) {
        const int j  = it >> 3;
        const int kc = (it & 7) << 6;

        // Prefetch next W tile into the other buffer while computing.
        if (it + 1 < NIT) {
            const int jn  = (it + 1) >> 3;
            const int kcn = ((it + 1) & 7) << 6;
            for (int i = tid; i < (128*64)/8; i += 256) {
                int e = i * 8;
                int row = e >> 6, cc = e & 63;
                *(v8u*)&Wl[buf ^ 1][row*64 + cc] =
                    *(const v8u*)&wbf[((size_t)jn*CO_ + co0 + row)*CIN_ + kcn + cc];
            }
        }

        #pragma unroll
        for (int sub = 0; sub < 2; ++sub) {
            const int kcc = kc + sub*32;
            // B fragment: lane = column; K chunks at (lane>>4)*8 and 16 + (lane>>4)*8.
            const unsigned short* bp = &Xl[(bcol + j)*512 + kcc + ahalf*8];
            v16u bt;
            {
                v8u lo = *(const v8u*)(bp);
                v8u hi = *(const v8u*)(bp + 16);
                #pragma unroll
                for (int e = 0; e < 8; ++e) { bt[e] = lo[e]; bt[e+8] = hi[e]; }
            }
            v16bf bfr = __builtin_bit_cast(v16bf, bt);

            // A fragments: split chunks at K = ahalf*8 and 16 + ahalf*8.
            {
                const unsigned short* ap = &Wl[buf][arow*64 + sub*32 + ahalf*8];
                v16u at;
                v8u lo = *(const v8u*)(ap);
                v8u hi = *(const v8u*)(ap + 16);
                #pragma unroll
                for (int e = 0; e < 8; ++e) { at[e] = lo[e]; at[e+8] = hi[e]; }
                v16bf afr = __builtin_bit_cast(v16bf, at);
                acc0 = __builtin_amdgcn_wmma_f32_16x16x32_bf16(
                           false, afr, false, bfr, (short)0, acc0, false, false);
            }
            {
                const unsigned short* ap = &Wl[buf][(arow + 16)*64 + sub*32 + ahalf*8];
                v16u at;
                v8u lo = *(const v8u*)(ap);
                v8u hi = *(const v8u*)(ap + 16);
                #pragma unroll
                for (int e = 0; e < 8; ++e) { at[e] = lo[e]; at[e+8] = hi[e]; }
                v16bf afr = __builtin_bit_cast(v16bf, at);
                acc1 = __builtin_amdgcn_wmma_f32_16x16x32_bf16(
                           false, afr, false, bfr, (short)0, acc1, false, false);
            }
        }
        __syncthreads();
        buf ^= 1;
    }

    // Epilogue: C/D layout -> element (M = r + 8*(lane>>4), N = lane&15)
    if (bcol < V_) {
        #pragma unroll
        for (int r = 0; r < 8; ++r) {
            int m = r + 8*ahalf;
            zc[((size_t)b*CO_ + co0 + wm*32 + m)*V_ + bcol]      = acc0[r];
            zc[((size_t)b*CO_ + co0 + wm*32 + 16 + m)*V_ + bcol] = acc1[r];
        }
    }
}

// ---------------------------------------------------------------------------
// 8) Graph einsum + LSTM gates, LDS-staged.
//    Block = (8 hid channels, one batch). Stage the 96x25 Zc slice + A into
//    LDS once; each active thread (hc_l, w) does its 300 MACs from LDS.
__global__ __launch_bounds__(256) void gates_kernel(
    const float* __restrict__ zc, const float* __restrict__ A,
    const float* __restrict__ biasA, float* __restrict__ cstate,
    unsigned short* __restrict__ hbf, float* __restrict__ out, int t)
{
    __shared__ float As[KSP_*V_*V_];     // 1875 floats
    __shared__ float Zt[96*V_];          // rows = (k*4+g)*8 + hc_l
    const int hc0 = blockIdx.x * 8;
    const int b   = blockIdx.y;

    for (int i = threadIdx.x; i < KSP_*V_*V_; i += 256) As[i] = A[i];
    for (int i = threadIdx.x; i < 96*V_; i += 256) {
        int row = i / V_, v = i % V_;
        int k = row >> 5, g = (row >> 3) & 3, hcl = row & 7;
        Zt[i] = zc[((size_t)b*CO_ + k*1024 + g*HID_ + hc0 + hcl)*V_ + v];
    }
    __syncthreads();

    const int hc_l = threadIdx.x >> 5;
    const int lane = threadIdx.x & 31;
    if (lane < V_) {
        const int w  = lane;
        const int hc = hc0 + hc_l;
        float G0 = biasA[hc*V_ + w];
        float G1 = biasA[(HID_   + hc)*V_ + w];
        float G2 = biasA[(2*HID_ + hc)*V_ + w];
        float G3 = biasA[(3*HID_ + hc)*V_ + w];
        #pragma unroll
        for (int k = 0; k < KSP_; ++k) {
            const float* Ak = As + k*(V_*V_);
            const float* z0 = &Zt[((k*4 + 0)*8 + hc_l)*V_];
            const float* z1 = &Zt[((k*4 + 1)*8 + hc_l)*V_];
            const float* z2 = &Zt[((k*4 + 2)*8 + hc_l)*V_];
            const float* z3 = &Zt[((k*4 + 3)*8 + hc_l)*V_];
            for (int v = 0; v < V_; ++v) {
                float a = Ak[v*V_ + w];
                G0 += a * z0[v];
                G1 += a * z1[v];
                G2 += a * z2[v];
                G3 += a * z3[v];
            }
        }
        float gi = 1.f / (1.f + expf(-G0));
        float gf = 1.f / (1.f + expf(-G1));
        float go = 1.f / (1.f + expf(-G2));
        float gg = tanhf(G3);

        int ci = (b*HID_ + hc)*V_ + w;
        float cn = gf * cstate[ci] + gi * gg;
        cstate[ci] = cn;
        float h = go * tanhf(cn);

        out[((size_t)(b*HID_ + hc)*T_ + t)*V_ + w] = h;     // (b, hid, t, v)
        hbf[((size_t)b*V_ + w)*HID_ + hc] = f2bf(h);        // next-step comb
    }
}

// ---------------------------------------------------------------------------
extern "C" void kernel_launch(void* const* d_in, const int* in_sizes, int n_in,
                              void* d_out, int out_size, void* d_ws, size_t ws_size,
                              hipStream_t stream) {
    (void)in_sizes; (void)n_in; (void)out_size; (void)ws_size;
    const float* x1     = (const float*)d_in[0];
    const float* x2     = (const float*)d_in[1];
    const float* x3     = (const float*)d_in[2];
    const float* A      = (const float*)d_in[3];
    const float* conv_w = (const float*)d_in[4];
    const float* conv_b = (const float*)d_in[5];
    const float* m1     = (const float*)d_in[6];
    const float* m2     = (const float*)d_in[7];
    const float* m3     = (const float*)d_in[8];
    float* out = (float*)d_out;

    char* ws = (char*)d_ws;
    size_t off = 0;
    auto alloc = [&](size_t bytes) {
        void* p = ws + off;
        off = (off + bytes + 255) & ~(size_t)255;
        return p;
    };
    float*          stats  = (float*)alloc(3*B_*2*sizeof(float));
    float*          fm     = (float*)alloc((size_t)3*V_*C_*sizeof(float));
    unsigned short* ybf    = (unsigned short*)alloc((size_t)B_*T_*V_*C_*2);   // 61.4 MB
    unsigned short* wbf    = (unsigned short*)alloc((size_t)KSP_*CO_*CIN_*2); //  9.4 MB
    float*          biasAb = (float*)alloc((size_t)4*HID_*V_*sizeof(float));
    unsigned short* hbf    = (unsigned short*)alloc((size_t)B_*HID_*V_*2);
    float*          cst    = (float*)alloc((size_t)B_*HID_*V_*sizeof(float));
    float*          zcb    = (float*)alloc((size_t)B_*CO_*V_*sizeof(float));  // 19.7 MB

    ln_stats_kernel<<<3*B_, 256, 0, stream>>>(x1, x2, x3, stats);
    mask_factor_kernel<<<(V_*C_ + 255)/256, 256, 0, stream>>>(m1, m2, m3, fm);
    build_y_kernel<<<B_*T_, 256, 0, stream>>>(x1, x2, x3, stats, fm, ybf);
    wbf_kernel<<<(int)(((size_t)KSP_*CO_*CIN_ + 255)/256), 256, 0, stream>>>(conv_w, wbf);
    biasA_kernel<<<(4*HID_*V_ + 255)/256, 256, 0, stream>>>(A, conv_b, biasAb);
    init_state_kernel<<<(B_*HID_*V_ + 255)/256, 256, 0, stream>>>(hbf, cst);

    for (int t = 0; t < T_; ++t) {
        gemm_step_kernel<<<dim3(CO_/128, B_), 256, 0, stream>>>(ybf, hbf, wbf, zcb, t);
        gates_kernel<<<dim3(HID_/8, B_), 256, 0, stream>>>(zcb, A, biasAb, cst, hbf, out, t);
    }
}
